// TemporalSAGE_14474039787721
// MI455X (gfx1250) — matrix operand order repaired and checked by reading, not compiled, not run on hardware
//
#include <hip/hip_runtime.h>
#include <hip/hip_bf16.h>
#include <math.h>

#define NNODES 50000
#define KNBR   16

typedef __attribute__((ext_vector_type(16))) __bf16 v16bf;
typedef __attribute__((ext_vector_type(8)))  __bf16 v8bf;
typedef __attribute__((ext_vector_type(8)))  float  v8f;

// ------------------------------------------------------------------
// Branch-free transcendentals. gfx1250 has hardware V_TANH_F32.
// ------------------------------------------------------------------
__device__ __forceinline__ float fast_tanh(float x) {
#if __has_builtin(__builtin_amdgcn_tanhf)
    return __builtin_amdgcn_tanhf(x);
#else
    // tanh(x) = sign(x) * (1 - e) / (1 + e),  e = exp(-2|x|)
    float e = __expf(-2.0f * fabsf(x));
#if __has_builtin(__builtin_amdgcn_rcpf)
    float t = (1.0f - e) * __builtin_amdgcn_rcpf(1.0f + e);
#else
    float t = (1.0f - e) / (1.0f + e);
#endif
    return copysignf(t, x);
#endif
}

__device__ __forceinline__ float sigm(float x) {
    // sigmoid(x) = 0.5 * (1 + tanh(x/2)) : single TRANS op
    return fmaf(0.5f, fast_tanh(0.5f * x), 0.5f);
}

// ------------------------------------------------------------------
// Repack [Wih;Whh] -> bf16, column-major over K=128 so each lane's
// WMMA B-fragment is one contiguous 32B load. Also fold bih+bhh.
// wcat[col*128 + k] : k<64 -> Wih[col,k], k>=64 -> Whh[col,k-64]
// ------------------------------------------------------------------
__global__ __launch_bounds__(256) void prep_wcat(
    const float* __restrict__ Wih, const float* __restrict__ Whh,
    const float* __restrict__ bih, const float* __restrict__ bhh,
    __bf16* __restrict__ wcat, float* __restrict__ bsum)
{
    int idx = blockIdx.x * 256 + threadIdx.x;
    if (idx < 256 * 128) {
        int col = idx >> 7;
        int k   = idx & 127;
        float v = (k < 64) ? Wih[col * 64 + k] : Whh[col * 64 + (k - 64)];
        wcat[idx] = (__bf16)v;
    }
    if (idx < 256) bsum[idx] = bih[idx] + bhh[idx];
}

// ------------------------------------------------------------------
// SAGEConv(lstm) for d=64 layers. One block = 16 nodes, 8 waves.
// Wave w computes gate columns [32w, 32w+32): 2 WMMA N-tiles,
// K=128 ( [m_t | h_{t-1}] concat ) in 4 chunks of 32.
// ------------------------------------------------------------------
template<int OUTF, bool RELU>
__global__ __launch_bounds__(256) void sage_layer64(
    const float* __restrict__ hin, const int* __restrict__ nbrs,
    const __bf16* __restrict__ wcat, const float* __restrict__ bsum,
    const float* __restrict__ Ws, const float* __restrict__ bs,
    const float* __restrict__ Wn, float* __restrict__ out)
{
    __shared__ __align__(16) __bf16 lds_m[16 * 72];   // m_t  bf16, padded stride
    __shared__ __align__(16) __bf16 lds_h[16 * 72];   // h_{t-1} bf16
    __shared__ float lds_c[16 * 64];                  // cell state fp32
    __shared__ float lds_hf[16 * 64];                 // h fp32 (final = hK)
    __shared__ float lds_self[16 * 64];               // layer input (self feats)
    __shared__ float lds_g[16 * 260];                 // gate pre-activations

    const int tid   = threadIdx.x;
    const int lane  = tid & 31;
    const int wv    = tid >> 5;
    const int node0 = blockIdx.x * 16;

    // load self features, zero h/c state
    {
        int n  = tid >> 4;
        int f0 = (tid & 15) << 2;
        float4 v = *(const float4*)(hin + (size_t)(node0 + n) * 64 + f0);
        *(float4*)(&lds_self[n * 64 + f0]) = v;
        #pragma unroll
        for (int i = 0; i < 4; ++i) {
            lds_c[n * 64 + f0 + i]  = 0.0f;
            lds_hf[n * 64 + f0 + i] = 0.0f;
            lds_h[n * 72 + f0 + i]  = (__bf16)0.0f;
        }
    }

    // persistent B fragments (contiguous K: lanes 0-15 K=0..15, 16-31 K=16..31)
    v16bf bfr[2][4];
    {
        int col  = lane & 15;
        int krow = (lane >> 4) << 4;              // 0 or 16
        #pragma unroll
        for (int ct = 0; ct < 2; ++ct) {
            const __bf16* base = wcat + (size_t)(wv * 32 + ct * 16 + col) * 128 + krow;
            #pragma unroll
            for (int kc = 0; kc < 4; ++kc)
                bfr[ct][kc] = *(const v16bf*)(base + kc * 32);
        }
    }

    // per-thread LSTM-cell constants: j is timestep- and rep-invariant
    const int ej = tid & 63;               // hidden unit
    const int en0 = tid >> 6;              // first node (reps add +4)
    const float bi_ = bsum[ej];
    const float bf_ = bsum[64  + ej];
    const float bg_ = bsum[128 + ej];
    const float bo_ = bsum[192 + ej];

    __syncthreads();

    for (int t = 0; t < KNBR; ++t) {
        // gather neighbor messages m_t -> LDS (bf16)
        {
            int n  = tid >> 4;
            int f0 = (tid & 15) << 2;
            int nb = nbrs[(node0 + n) * KNBR + t];
            float4 v = *(const float4*)(hin + (size_t)nb * 64 + f0);
            lds_m[n * 72 + f0 + 0] = (__bf16)v.x;
            lds_m[n * 72 + f0 + 1] = (__bf16)v.y;
            lds_m[n * 72 + f0 + 2] = (__bf16)v.z;
            lds_m[n * 72 + f0 + 3] = (__bf16)v.w;
            if (f0 == 0 && t + 1 < KNBR) {        // prefetch next timestep's row
                int nb2 = nbrs[(node0 + n) * KNBR + t + 1];
                __builtin_prefetch(hin + (size_t)nb2 * 64, 0, 1);
            }
        }
        __syncthreads();

        // A fragments (contiguous K, matching B): lanes 0-15 K base 0, 16-31 K base 16;
        // elements 0..15 -> K base..base+15
        v16bf afr[4];
        {
            int r  = lane & 15;
            int kb = (lane >> 4) << 4;            // 0 or 16
            const __bf16* mrow = lds_m + r * 72;
            const __bf16* hrow = lds_h + r * 72;
            #pragma unroll
            for (int kc = 0; kc < 4; ++kc) {
                const __bf16* src = (kc < 2) ? (mrow + kc * 32 + kb)
                                             : (hrow + (kc - 2) * 32 + kb);
                v8bf lo = *(const v8bf*)(src);
                v8bf hi = *(const v8bf*)(src + 8);
                #pragma unroll
                for (int e = 0; e < 8; ++e) { afr[kc][e] = lo[e]; afr[kc][e + 8] = hi[e]; }
            }
        }

        v8f acc0 = {}; v8f acc1 = {};
        #pragma unroll
        for (int kc = 0; kc < 4; ++kc) {
            acc0 = __builtin_amdgcn_wmma_f32_16x16x32_bf16(false, afr[kc], false, bfr[0][kc],
                                                           (short)0, acc0, false, false);
            acc1 = __builtin_amdgcn_wmma_f32_16x16x32_bf16(false, afr[kc], false, bfr[1][kc],
                                                           (short)0, acc1, false, false);
        }

        // scatter D tiles (VGPR r: lanes0-15 M=r, lanes16-31 M=r+8)
        {
            int l  = lane & 15;
            int hi = lane >> 4;
            #pragma unroll
            for (int r = 0; r < 8; ++r) {
                int row = r + (hi << 3);
                lds_g[row * 260 + wv * 32 + l]      = acc0[r];
                lds_g[row * 260 + wv * 32 + 16 + l] = acc1[r];
            }
        }
        __syncthreads();

        // LSTM cell update (torch gate order i,f,g,o), fp32, branch-free TRANS
        #pragma unroll
        for (int rep = 0; rep < 4; ++rep) {
            int n = en0 + rep * 4;
            float gi = lds_g[n * 260 + ej]       + bi_;
            float gf = lds_g[n * 260 + 64 + ej]  + bf_;
            float gg = lds_g[n * 260 + 128 + ej] + bg_;
            float go = lds_g[n * 260 + 192 + ej] + bo_;
            float c  = lds_c[n * 64 + ej];
            c = sigm(gf) * c + sigm(gi) * fast_tanh(gg);
            float h = sigm(go) * fast_tanh(c);
            lds_c[n * 64 + ej]  = c;
            lds_hf[n * 64 + ej] = h;
            lds_h[n * 72 + ej]  = (__bf16)h;
        }
        // no barrier needed here: these LDS writes are ordered against the
        // next iteration's reads by the next post-gather __syncthreads()
    }
    __syncthreads();

    // out = self @ Ws^T + bs + hK @ Wn^T  (tiny: VALU)
    if (OUTF == 64) {
        #pragma unroll
        for (int rep = 0; rep < 4; ++rep) {
            int n = en0 + rep * 4, jo = ej;
            float acc = bs[jo];
            #pragma unroll 8
            for (int k = 0; k < 64; ++k)
                acc += lds_self[n * 64 + k] * Ws[jo * 64 + k]
                     + lds_hf[n * 64 + k]   * Wn[jo * 64 + k];
            if (RELU) acc = fmaxf(acc, 0.0f);
            out[(size_t)(node0 + n) * 64 + jo] = acc;
        }
    } else {
        if (tid < 16) {
            int n = tid;
            float acc = bs[0];
            #pragma unroll 8
            for (int k = 0; k < 64; ++k)
                acc += lds_self[n * 64 + k] * Ws[k] + lds_hf[n * 64 + k] * Wn[k];
            out[node0 + n] = acc;
        }
    }
}

// ------------------------------------------------------------------
// Layer 1: d=11 LSTM — too small for WMMA tiles; one thread per node,
// weights cached in LDS. <5% of total FLOPs.
// ------------------------------------------------------------------
__global__ __launch_bounds__(256) void sage_layer1(
    const float* __restrict__ x, const int* __restrict__ nbrs,
    const float* __restrict__ Wih, const float* __restrict__ Whh,
    const float* __restrict__ bih, const float* __restrict__ bhh,
    const float* __restrict__ Ws, const float* __restrict__ bs,
    const float* __restrict__ Wn, float* __restrict__ out)
{
    __shared__ float s_Wih[44 * 11];
    __shared__ float s_Whh[44 * 11];
    __shared__ float s_b[44];
    __shared__ float s_Ws[64 * 11];
    __shared__ float s_bs[64];
    __shared__ float s_Wn[64 * 11];

    for (int i = threadIdx.x; i < 44 * 11; i += 256) { s_Wih[i] = Wih[i]; s_Whh[i] = Whh[i]; }
    for (int i = threadIdx.x; i < 44;      i += 256) s_b[i] = bih[i] + bhh[i];
    for (int i = threadIdx.x; i < 64 * 11; i += 256) { s_Ws[i] = Ws[i]; s_Wn[i] = Wn[i]; }
    for (int i = threadIdx.x; i < 64;      i += 256) s_bs[i] = bs[i];
    __syncthreads();

    int node = blockIdx.x * 256 + threadIdx.x;
    if (node >= NNODES) return;

    float xs[11], h[11], c[11];
    #pragma unroll
    for (int k = 0; k < 11; ++k) { xs[k] = x[(size_t)node * 11 + k]; h[k] = 0.f; c[k] = 0.f; }

    for (int t = 0; t < KNBR; ++t) {
        int nb = nbrs[node * KNBR + t];
        float m[11];
        #pragma unroll
        for (int k = 0; k < 11; ++k) m[k] = x[(size_t)nb * 11 + k];
        float hn[11];
        #pragma unroll
        for (int j = 0; j < 11; ++j) {
            float gi = s_b[j], gf = s_b[11 + j], gg = s_b[22 + j], go = s_b[33 + j];
            #pragma unroll
            for (int k = 0; k < 11; ++k) {
                gi += m[k] * s_Wih[j * 11 + k]        + h[k] * s_Whh[j * 11 + k];
                gf += m[k] * s_Wih[(11 + j) * 11 + k] + h[k] * s_Whh[(11 + j) * 11 + k];
                gg += m[k] * s_Wih[(22 + j) * 11 + k] + h[k] * s_Whh[(22 + j) * 11 + k];
                go += m[k] * s_Wih[(33 + j) * 11 + k] + h[k] * s_Whh[(33 + j) * 11 + k];
            }
            c[j] = sigm(gf) * c[j] + sigm(gi) * fast_tanh(gg);
            hn[j] = sigm(go) * fast_tanh(c[j]);
        }
        #pragma unroll
        for (int j = 0; j < 11; ++j) h[j] = hn[j];
    }

    for (int jo = 0; jo < 64; ++jo) {
        float acc = s_bs[jo];
        #pragma unroll
        for (int k = 0; k < 11; ++k)
            acc += xs[k] * s_Ws[jo * 11 + k] + h[k] * s_Wn[jo * 11 + k];
        out[(size_t)node * 64 + jo] = fmaxf(acc, 0.0f);
    }
}

// ------------------------------------------------------------------
// GRUCell (r,z,n order) — ~3% of FLOPs, VALU. 4 nodes per block.
// ------------------------------------------------------------------
__global__ __launch_bounds__(256) void gru_kernel(
    const float* __restrict__ x, const float* __restrict__ hprev,
    const float* __restrict__ Wih, const float* __restrict__ Whh,
    const float* __restrict__ bih, const float* __restrict__ bhh,
    float* __restrict__ out)
{
    __shared__ float xs[4 * 64];
    __shared__ float hs[4 * 64];
    const int tid   = threadIdx.x;
    const int node0 = blockIdx.x * 4;
    xs[tid] = x[(size_t)node0 * 64 + tid];
    hs[tid] = hprev[(size_t)node0 * 64 + tid];
    __syncthreads();

    int n = tid >> 6, j = tid & 63;
    float ir = bih[j], iz = bih[64 + j], in_ = bih[128 + j];
    float hr = bhh[j], hz = bhh[64 + j], hn  = bhh[128 + j];
    const float* xv = xs + n * 64;
    const float* hv = hs + n * 64;
    #pragma unroll 4
    for (int k = 0; k < 64; ++k) {
        float xk = xv[k], hk = hv[k];
        ir  += xk * Wih[j * 64 + k];
        iz  += xk * Wih[(64 + j) * 64 + k];
        in_ += xk * Wih[(128 + j) * 64 + k];
        hr  += hk * Whh[j * 64 + k];
        hz  += hk * Whh[(64 + j) * 64 + k];
        hn  += hk * Whh[(128 + j) * 64 + k];
    }
    float r  = sigm(ir + hr);
    float z  = sigm(iz + hz);
    float nn = fast_tanh(in_ + r * hn);
    out[(size_t)node0 * 64 + tid] = (1.0f - z) * nn + z * hv[j];
}

// ------------------------------------------------------------------
extern "C" void kernel_launch(void* const* d_in, const int* in_sizes, int n_in,
                              void* d_out, int out_size, void* d_ws, size_t ws_size,
                              hipStream_t stream)
{
    (void)in_sizes; (void)n_in; (void)out_size; (void)ws_size;

    const float* x     = (const float*)d_in[0];
    const int*   nbrs  = (const int*)d_in[1];
    const float* hprev = (const float*)d_in[2];
    // l1: 3..9, l2: 10..16, l3: 17..23, l4: 24..30, gru: 31..34

    float*  bufA  = (float*)d_ws;                           // [N,64]
    float*  bufB  = bufA + (size_t)NNODES * 64;             // [N,64]
    __bf16* wcat2 = (__bf16*)(bufB + (size_t)NNODES * 64);  // 3x [256,128] bf16
    __bf16* wcat3 = wcat2 + 256 * 128;
    __bf16* wcat4 = wcat3 + 256 * 128;
    float*  bsum2 = (float*)(wcat4 + 256 * 128);
    float*  bsum3 = bsum2 + 256;
    float*  bsum4 = bsum3 + 256;

    dim3 b256(256);

    prep_wcat<<<128, b256, 0, stream>>>((const float*)d_in[10], (const float*)d_in[11],
                                        (const float*)d_in[12], (const float*)d_in[13],
                                        wcat2, bsum2);
    prep_wcat<<<128, b256, 0, stream>>>((const float*)d_in[17], (const float*)d_in[18],
                                        (const float*)d_in[19], (const float*)d_in[20],
                                        wcat3, bsum3);
    prep_wcat<<<128, b256, 0, stream>>>((const float*)d_in[24], (const float*)d_in[25],
                                        (const float*)d_in[26], (const float*)d_in[27],
                                        wcat4, bsum4);

    sage_layer1<<<(NNODES + 255) / 256, b256, 0, stream>>>(x, nbrs,
        (const float*)d_in[3], (const float*)d_in[4], (const float*)d_in[5], (const float*)d_in[6],
        (const float*)d_in[7], (const float*)d_in[8], (const float*)d_in[9], bufA);

    sage_layer64<64, true><<<NNODES / 16, b256, 0, stream>>>(bufA, nbrs, wcat2, bsum2,
        (const float*)d_in[14], (const float*)d_in[15], (const float*)d_in[16], bufB);

    sage_layer64<64, true><<<NNODES / 16, b256, 0, stream>>>(bufB, nbrs, wcat3, bsum3,
        (const float*)d_in[21], (const float*)d_in[22], (const float*)d_in[23], bufA);

    gru_kernel<<<NNODES / 4, b256, 0, stream>>>(bufA, hprev,
        (const float*)d_in[31], (const float*)d_in[32], (const float*)d_in[33],
        (const float*)d_in[34], bufB);

    sage_layer64<1, false><<<NNODES / 16, b256, 0, stream>>>(bufB, nbrs, wcat4, bsum4,
        (const float*)d_in[28], (const float*)d_in[29], (const float*)d_in[30], (float*)d_out);
}
